// DConv2d_66236985639657
// MI455X (gfx1250) — matrix operand order repaired and checked by reading, not compiled, not run on hardware
//
#include <hip/hip_runtime.h>

typedef __attribute__((ext_vector_type(16))) _Float16 v16h;
typedef __attribute__((ext_vector_type(8)))  _Float16 v8h;
typedef __attribute__((ext_vector_type(8)))  float    v8f;

#define Bn    8
#define CINn  32
#define COUTn 32
#define Hn    512
#define Wn    512

#define HB    4            // output rows per workgroup
#define RS    (HB + 2)     // staged rows (halo)
#define TW    128          // output pixels (w) per workgroup (8 waves x 16)
#define PXn   (TW + 2)     // 130 staged pixels per row
#define XSTR  40           // halves per pixel in sx (32 data + 8 pad) -> 80B stride, bank-conflict-free
#define ASTR  16           // halves per A-fragment slot (32B, aligned)

#define PACKA_U32 (18 * 32 * 8)              // packed A-fragment table size in dwords
#define PACKA_BYTES (PACKA_U32 * 4)

// ---------------------------------------------------------------------------
// One-time weight prepack: per-lane WMMA A fragments for 9 taps x 2 cout blocks,
// using the ISA 16-bit A 16x32 VGPR layout:
//   lane L, VGPR v: K = (v/4)*16 + (L>=16 ? 8 : 0) + (v%4)*2 + {0,1}; M = L%16
// ---------------------------------------------------------------------------
__global__ void pack_weights_kernel(const float* __restrict__ weight,
                                    unsigned* __restrict__ packedA)
{
  for (int idx = threadIdx.x; idx < PACKA_U32; idx += 256) {
    const int q    = idx & 7;           // VGPR index v
    const int L    = (idx >> 3) & 31;   // lane
    const int s    = idx >> 8;          // slot = tap*2 + coutblock
    const int tap  = s >> 1;
    const int blkm = s & 1;
    const int ti = tap / 3, tj = tap % 3;
    const int base = (L & 16) ? 8 : 0;
    const int K0   = ((q >> 2) << 4) + base + ((q & 3) << 1);
    const int co   = blkm * 16 + (L & 15);
    union { _Float16 h2[2]; unsigned u; } pk;
    pk.h2[0] = (_Float16)weight[((co * CINn + K0    ) * 3 + ti) * 3 + tj];
    pk.h2[1] = (_Float16)weight[((co * CINn + K0 + 1) * 3 + ti) * 3 + tj];
    packedA[idx] = pk.u;
  }
}

// ---------------------------------------------------------------------------
// Main kernel: each block computes HB x TW output pixels for all 32 couts.
// ---------------------------------------------------------------------------
__global__ __launch_bounds__(256, 2)
void dconv_wmma_kernel(const float* __restrict__ x,
                       const float* __restrict__ depth,
                       const float* __restrict__ weight,
                       const float* __restrict__ bias,
                       const unsigned* __restrict__ packedA,   // may be null -> in-block gather
                       float* __restrict__ out)
{
  __shared__ __attribute__((aligned(32))) _Float16 sa[18 * 32 * ASTR]; // A frags (18.4 KB)
  __shared__ __attribute__((aligned(32))) _Float16 sx[RS * PXn * XSTR]; // x slab f16 (62.4 KB)
  __shared__ float sd[RS * PXn];                                        // depth slab (3.1 KB)

  const int tid  = threadIdx.x;
  const int lane = tid & 31;
  const int wave = tid >> 5;

  const int blk  = blockIdx.x;
  const int wblk = blk & 3;             // Wn/TW = 4
  const int hblk = (blk >> 2) & 127;    // Hn/HB = 128
  const int b    = blk >> 9;
  const int w0   = wblk * TW;
  const int h0   = hblk * HB;

  // ---- stage x slab (fp32 -> f16), layout sx[row][pixel][cin], zero-padded halo ----
  for (int idx = tid; idx < 16 * RS * PXn; idx += 256) {
    const int wp    = idx % PXn;
    const int t     = idx / PXn;
    const int row   = t % RS;           // h0-1 .. h0+HB
    const int cpair = t / RS;           // cin pair 0..15
    const int gh = h0 - 1 + row;
    const int gw = w0 - 1 + wp;
    float f0 = 0.0f, f1 = 0.0f;
    if (gh >= 0 && gh < Hn && gw >= 0 && gw < Wn) {
      const float* p = x + (((size_t)b * CINn + 2 * cpair) * Hn + gh) * Wn + gw;
      f0 = p[0];
      f1 = p[(size_t)Hn * Wn];
    }
    union { _Float16 h2[2]; unsigned u; } pk;
    pk.h2[0] = (_Float16)f0;
    pk.h2[1] = (_Float16)f1;
    *(unsigned*)&sx[(row * PXn + wp) * XSTR + 2 * cpair] = pk.u;
  }

  // ---- stage depth slab (fp32), zero-padded halo ----
  for (int idx = tid; idx < RS * PXn; idx += 256) {
    const int wp  = idx % PXn;
    const int row = idx / PXn;
    const int gh = h0 - 1 + row;
    const int gw = w0 - 1 + wp;
    float f = 0.0f;
    if (gh >= 0 && gh < Hn && gw >= 0 && gw < Wn)
      f = depth[((size_t)b * Hn + gh) * Wn + gw];
    sd[row * PXn + wp] = f;
  }

  // ---- stage A fragments: coalesced copy from prepacked table (fast path) ----
  if (packedA != nullptr) {
    unsigned* sau = (unsigned*)sa;
    for (int idx = tid; idx < PACKA_U32; idx += 256)
      sau[idx] = packedA[idx];
  } else {
    for (int idx = tid; idx < PACKA_U32; idx += 256) {
      const int q    = idx & 7;
      const int L    = (idx >> 3) & 31;
      const int s    = idx >> 8;
      const int tap  = s >> 1;
      const int blkm = s & 1;
      const int ti = tap / 3, tj = tap % 3;
      const int base = (L & 16) ? 8 : 0;
      const int K0   = ((q >> 2) << 4) + base + ((q & 3) << 1);
      const int co   = blkm * 16 + (L & 15);
      union { _Float16 h2[2]; unsigned u; } pk;
      pk.h2[0] = (_Float16)weight[((co * CINn + K0    ) * 3 + ti) * 3 + tj];
      pk.h2[1] = (_Float16)weight[((co * CINn + K0 + 1) * 3 + ti) * 3 + tj];
      *(unsigned*)&sa[(s * 32 + L) * ASTR + 2 * q] = pk.u;
    }
  }

  __syncthreads();

  // ---- compute: each wave owns one 16-pixel w-segment across all HB rows ----
  const int n  = lane & 15;                 // column N = pixel within tile
  const int pl = 1 + wave * 16 + n;         // pixel index in staged slab
  const int kh = (lane & 16) >> 1;          // cin base for B fragment (0 or 8)
  const int mb = (lane & 16) ? 8 : 0;       // C/D row base

  v8f cb0, cb1;                             // bias templates
#pragma unroll
  for (int v = 0; v < 8; ++v) {
    cb0[v] = bias[mb + v];
    cb1[v] = bias[16 + mb + v];
  }

  const int wout = w0 + wave * 16 + n;

  for (int r = 0; r < HB; ++r) {
    const float dc = sd[(1 + r) * PXn + pl];    // center depth for this row
    v8f c0 = cb0, c1 = cb1;

#pragma unroll
    for (int i = 0; i < 3; ++i) {
#pragma unroll
      for (int j = 0; j < 3; ++j) {
        const int pp  = pl + j - 1;
        const int srow = r + i;                  // staged row for this tap
        const float dsv = sd[srow * PXn + pp];
        const float mk  = __expf(-fabsf(dc - dsv));
        const _Float16 mh = (_Float16)mk;

        // B fragment: f16 cin kh..kh+7 and 16+kh..16+kh+7 for this lane's pixel
        const _Float16* bp = &sx[(srow * PXn + pp) * XSTR + kh];
        v8h blo = *(const v8h*)bp;
        v8h bhi = *(const v8h*)(bp + 16);
        v16h bfrag = __builtin_shufflevector(blo, bhi,
                       0,1,2,3,4,5,6,7,8,9,10,11,12,13,14,15);
        bfrag = bfrag * mh;                      // packed f16 mask multiply

        const int s = (i * 3 + j) * 2;
        const v16h a0 = *(const v16h*)&sa[( s      * 32 + lane) * ASTR];
        const v16h a1 = *(const v16h*)&sa[((s + 1) * 32 + lane) * ASTR];

        c0 = __builtin_amdgcn_wmma_f32_16x16x32_f16(false, a0, false, bfrag,
                                                    (short)0, c0, false, false);
        c1 = __builtin_amdgcn_wmma_f32_16x16x32_f16(false, a1, false, bfrag,
                                                    (short)0, c1, false, false);
      }
    }

    // ---- store row h0+r: lane = column w, VGPR v = cout row (C/D layout) ----
    float* o0 = out + (((size_t)b * COUTn + mb) * Hn + (h0 + r)) * Wn + wout;
#pragma unroll
    for (int v = 0; v < 8; ++v) {
      o0[(size_t)v * Hn * Wn]        = c0[v];
      o0[(size_t)(16 + v) * Hn * Wn] = c1[v];
    }
  }
}

extern "C" void kernel_launch(void* const* d_in, const int* in_sizes, int n_in,
                              void* d_out, int out_size, void* d_ws, size_t ws_size,
                              hipStream_t stream) {
  const float* x      = (const float*)d_in[0];
  const float* depth  = (const float*)d_in[1];
  const float* weight = (const float*)d_in[2];
  const float* bias   = (const float*)d_in[3];
  float* out = (float*)d_out;

  unsigned* packedA = nullptr;
  if (ws_size >= (size_t)PACKA_BYTES && d_ws != nullptr) {
    packedA = (unsigned*)d_ws;
    hipLaunchKernelGGL(pack_weights_kernel, dim3(1), dim3(256), 0, stream,
                       weight, packedA);
  }

  dim3 grid(Bn * (Hn / HB) * (Wn / TW));   // 8 * 128 * 4 = 4096 workgroups
  dim3 block(256);                          // 8 wave32 waves
  hipLaunchKernelGGL(dconv_wmma_kernel, grid, block, 0, stream,
                     x, depth, weight, bias, packedA, out);
}
